// sparseESN_4655744549591
// MI455X (gfx1250) — matrix operand order repaired and checked by reading, compiled, not run on hardware
//
#include <hip/hip_runtime.h>
#include <hip/hip_bf16.h>
#include <math.h>

// ---------------- problem constants (from reference) ----------------
#define RES   4096
#define NF    512
#define BATCH 256
#define TSTEP 64
// damping a = 0.3 -> x_new = 0.7*x + 0.3*tanh(pre)

typedef __attribute__((ext_vector_type(16))) __bf16        v16bf;
typedef __attribute__((ext_vector_type(8)))  float         v8f;
typedef __attribute__((ext_vector_type(4)))  unsigned int  v4u;

union FragAB { v16bf bf; v4u u4[2]; };   // 32 bytes: 16 bf16 = one WMMA A or B operand

__device__ __forceinline__ unsigned short f2bf(float f) {
  unsigned u = __float_as_uint(f);
  unsigned r = u + 0x7FFFu + ((u >> 16) & 1u);   // round-to-nearest-even
  return (unsigned short)(r >> 16);
}

// ---------------------------------------------------------------------------
// Convert W (f32 row-major [RES][RES], pre = x @ W so W[k][n]) into bf16 packed
// in exact WMMA B-fragment layout. Tile = 32(K) x 16(N). Contiguous k order (this code's choice, A loads match):
//   lane = khalf*16 + n  (khalf = K-half 0/1, n = column 0..15)
//   dword v (0..7) holds K = khalf*16 + 2v (low half) and +1 (high half)
// Tile stored as 256 contiguous dwords at dword offset lane*8 + v.
// Tile order: nTile-major, kTile inner  -> tileIdx = nTile*128 + kTile.
// tid = (((nTile*128)+kTile)*32 + lane)*8 + v  == destination dword index.
// ---------------------------------------------------------------------------
__global__ void convert_W_kernel(const float* __restrict__ W,
                                 unsigned int* __restrict__ Wp) {
  unsigned tid = blockIdx.x * blockDim.x + threadIdx.x;   // < 8388608
  unsigned v     = tid & 7u;
  unsigned lane  = (tid >> 3) & 31u;
  unsigned kTile = (tid >> 8) & 127u;
  unsigned nTile = tid >> 15;
  unsigned n     = nTile * 16u + (lane & 15u);
  unsigned khalf = lane >> 4;
  unsigned k0    = kTile * 32u + khalf * 16u + 2u * v;
  unsigned short lo = f2bf(W[(size_t)k0 * RES + n]);
  unsigned short hi = f2bf(W[(size_t)(k0 + 1u) * RES + n]);
  Wp[tid] = (unsigned)lo | ((unsigned)hi << 16);
}

// Same for Win rows 1..512 (row 0 is the bias, handled in the epilogue).
// kTiles = 16 -> tileIdx = nTile*16 + kTile ; tid is the dest dword index.
__global__ void convert_Win_kernel(const float* __restrict__ Win,
                                   unsigned int* __restrict__ WinP) {
  unsigned tid = blockIdx.x * blockDim.x + threadIdx.x;   // < 1048576
  unsigned v     = tid & 7u;
  unsigned lane  = (tid >> 3) & 31u;
  unsigned kTile = (tid >> 8) & 15u;
  unsigned nTile = tid >> 12;
  unsigned n     = nTile * 16u + (lane & 15u);
  unsigned khalf = lane >> 4;
  unsigned k0    = kTile * 32u + khalf * 16u + 2u * v;    // 0..510
  unsigned short lo = f2bf(Win[(size_t)(1u + k0) * RES + n]);
  unsigned short hi = f2bf(Win[(size_t)(2u + k0) * RES + n]);
  WinP[tid] = (unsigned)lo | ((unsigned)hi << 16);
}

// u: plain f32 -> bf16, row-major (A fragments load row-major directly).
__global__ void convert_u_kernel(const float* __restrict__ u,
                                 unsigned int* __restrict__ ub) {
  unsigned tid = blockIdx.x * blockDim.x + threadIdx.x;   // < 4194304
  ub[tid] = (unsigned)f2bf(u[2u * tid]) | ((unsigned)f2bf(u[2u * tid + 1u]) << 16);
}

// x0 -> f32 master copy + bf16 shadow.
__global__ void init_x_kernel(const float* __restrict__ x0,
                              float* __restrict__ xf,
                              unsigned int* __restrict__ xb) {
  unsigned tid = blockIdx.x * blockDim.x + threadIdx.x;   // < 524288
  float a = x0[2u * tid], b = x0[2u * tid + 1u];
  xf[2u * tid] = a; xf[2u * tid + 1u] = b;
  xb[tid] = (unsigned)f2bf(a) | ((unsigned)f2bf(b) << 16);
}

// ---------------------------------------------------------------------------
// One ESN step. Grid: 128 blocks (one per 32-column group), 4 waves/block.
// Wave w computes rows [w*64, w*64+64) x 32 cols: 4x2 tiles of 16x16.
//   pre = x_bf16 @ W_bf16  (K=4096, 128 k-tiles)
//       + u_t_bf16 @ Win_bf16[1:513] (K=512, 16 k-tiles)
//       + Win[0,:]                      (bias, f32)
//   x_new = 0.7*x_old_f32 + 0.3*tanh(pre); write f32 + bf16 shadow.
// ---------------------------------------------------------------------------
__global__ __launch_bounds__(128) void esn_step_kernel(
    const unsigned int* __restrict__ Wp,      // swizzled bf16 W
    const unsigned int* __restrict__ WinP,    // swizzled bf16 Win[1:]
    const unsigned short* __restrict__ ub,    // bf16 u, row-major [T][B][NF]
    const float* __restrict__ Win_f32,        // raw Win (for bias row 0)
    const float* __restrict__ x_old_f32,
    const unsigned short* __restrict__ x_old_bf16,
    float* __restrict__ x_new_f32,
    unsigned short* __restrict__ x_new_bf16,
    int t) {
  const int lane  = threadIdx.x & 31;
  const int wid   = threadIdx.x >> 5;
  const int khalf = lane >> 4;      // K-half selector (ISA 16-bit A/B layouts)
  const int nlo   = lane & 15;      // column within 16-wide tile / A row
  const int mBase = wid * 64;       // this wave's 64 rows
  const int ntG0  = blockIdx.x * 2; // this wave's first N-tile (global)

  v8f acc[4][2];
#pragma unroll
  for (int mt = 0; mt < 4; ++mt)
#pragma unroll
    for (int nt = 0; nt < 2; ++nt) acc[mt][nt] = (v8f){0.f,0.f,0.f,0.f,0.f,0.f,0.f,0.f};

  // ---- phase 1: x @ W, K = 4096 (128 k-tiles of 32) ----
  for (int kt = 0; kt < 128; ++kt) {
    FragAB b[2];
#pragma unroll
    for (int nt = 0; nt < 2; ++nt) {
      const unsigned* p = Wp + ((((size_t)(ntG0 + nt)) * 128 + kt) << 8) + (lane << 3);
      b[nt].u4[0] = *(const v4u*)p;
      b[nt].u4[1] = *(const v4u*)(p + 4);
    }
    FragAB a[4];
#pragma unroll
    for (int mt = 0; mt < 4; ++mt) {
      const unsigned short* ap =
          x_old_bf16 + (size_t)(mBase + mt * 16 + nlo) * RES + kt * 32 + khalf * 16;
      a[mt].u4[0] = *(const v4u*)ap;
      a[mt].u4[1] = *(const v4u*)(ap + 8);
    }
#pragma unroll
    for (int mt = 0; mt < 4; ++mt)
#pragma unroll
      for (int nt = 0; nt < 2; ++nt)
        acc[mt][nt] = __builtin_amdgcn_wmma_f32_16x16x32_bf16(
            false, a[mt].bf, false, b[nt].bf, (short)0, acc[mt][nt], false, false);
  }

  // ---- phase 2: u_t @ Win[1:], K = 512 (16 k-tiles of 32) ----
  const unsigned short* uBase = ub + (size_t)t * BATCH * NF;
  for (int kt = 0; kt < 16; ++kt) {
    FragAB b[2];
#pragma unroll
    for (int nt = 0; nt < 2; ++nt) {
      const unsigned* p = WinP + ((((size_t)(ntG0 + nt)) * 16 + kt) << 8) + (lane << 3);
      b[nt].u4[0] = *(const v4u*)p;
      b[nt].u4[1] = *(const v4u*)(p + 4);
    }
    FragAB a[4];
#pragma unroll
    for (int mt = 0; mt < 4; ++mt) {
      const unsigned short* ap =
          uBase + (size_t)(mBase + mt * 16 + nlo) * NF + kt * 32 + khalf * 16;
      a[mt].u4[0] = *(const v4u*)ap;
      a[mt].u4[1] = *(const v4u*)(ap + 8);
    }
#pragma unroll
    for (int mt = 0; mt < 4; ++mt)
#pragma unroll
      for (int nt = 0; nt < 2; ++nt)
        acc[mt][nt] = __builtin_amdgcn_wmma_f32_16x16x32_bf16(
            false, a[mt].bf, false, b[nt].bf, (short)0, acc[mt][nt], false, false);
  }

  // ---- epilogue: bias + tanh + damping; write f32 + bf16 shadow ----
#pragma unroll
  for (int nt = 0; nt < 2; ++nt) {
    const int n = (ntG0 + nt) * 16 + nlo;
    const float bias = Win_f32[n];           // Win[0][n]
#pragma unroll
    for (int mt = 0; mt < 4; ++mt) {
#pragma unroll
      for (int r = 0; r < 8; ++r) {
        // C/D layout: lanes 0-15 -> M=r, lanes 16-31 -> M=r+8; N = lane&15
        const int m = mBase + mt * 16 + r + khalf * 8;
        const float pre = acc[mt][nt][r] + bias;
        const float xo  = x_old_f32[(size_t)m * RES + n];
        const float xn  = 0.7f * xo + 0.3f * tanhf(pre);
        x_new_f32[(size_t)m * RES + n]  = xn;
        x_new_bf16[(size_t)m * RES + n] = f2bf(xn);
      }
    }
  }
}

// ---------------------------------------------------------------------------
// Workspace layout (needs 64 MB):
//   [0,      32MB) Wp      bf16 swizzled W
//   [32MB,   36MB) WinP    bf16 swizzled Win[1:513]
//   [36MB,   52MB) u_bf16  row-major
//   [52MB,   60MB) x_f32   double buffer (2 x 4MB)
//   [60MB,   64MB) x_bf16  double buffer (2 x 2MB)
// ---------------------------------------------------------------------------
extern "C" void kernel_launch(void* const* d_in, const int* in_sizes, int n_in,
                              void* d_out, int out_size, void* d_ws, size_t ws_size,
                              hipStream_t stream) {
  (void)in_sizes; (void)n_in; (void)out_size; (void)ws_size;
  const float* u   = (const float*)d_in[0];   // [64][256][512]
  const float* x0  = (const float*)d_in[1];   // [256][4096]
  const float* Win = (const float*)d_in[2];   // [513][4096]
  const float* W   = (const float*)d_in[3];   // [4096][4096]

  char* ws = (char*)d_ws;
  unsigned int*   Wp   = (unsigned int*)(ws + 0);
  unsigned int*   WinP = (unsigned int*)(ws + (size_t)32 * 1024 * 1024);
  unsigned int*   uB   = (unsigned int*)(ws + (size_t)36 * 1024 * 1024);
  float*          xf[2];
  unsigned int*   xb[2];
  xf[0] = (float*)(ws + (size_t)52 * 1024 * 1024);
  xf[1] = (float*)(ws + (size_t)56 * 1024 * 1024);
  xb[0] = (unsigned int*)(ws + (size_t)60 * 1024 * 1024);
  xb[1] = (unsigned int*)(ws + (size_t)62 * 1024 * 1024);

  convert_W_kernel  <<<32768, 256, 0, stream>>>(W, Wp);
  convert_Win_kernel<<< 4096, 256, 0, stream>>>(Win, WinP);
  convert_u_kernel  <<<16384, 256, 0, stream>>>(u, uB);
  init_x_kernel     <<< 2048, 256, 0, stream>>>(x0, xf[0], xb[0]);

  for (int t = 0; t < TSTEP; ++t) {
    const int cur = t & 1, nxt = cur ^ 1;
    float* outF = (t == TSTEP - 1) ? (float*)d_out : xf[nxt];
    esn_step_kernel<<<128, 128, 0, stream>>>(
        Wp, WinP, (const unsigned short*)uB, Win,
        xf[cur], (const unsigned short*)xb[cur],
        outF, (unsigned short*)xb[nxt], t);
  }
}
